// SimpleMHA2D_22436909154659
// MI455X (gfx1250) — matrix operand order repaired and checked by reading, compiled, not run on hardware
//
#include <hip/hip_runtime.h>
#include <hip/hip_bf16.h>
#include <math.h>

typedef __attribute__((ext_vector_type(16))) __bf16 v16bf;
typedef __attribute__((ext_vector_type(8)))  __bf16 v8bf;
typedef __attribute__((ext_vector_type(2)))  __bf16 v2bf;
typedef __attribute__((ext_vector_type(8)))  float  v8f;
typedef __attribute__((ext_vector_type(4)))  float  v4f;
typedef __attribute__((ext_vector_type(2)))  float  v2f;

#define B_DIM 32
#define S_DIM 4096
#define C_DIM 512
#define NH    16
#define KD    64
#define VD    64
#define NCOL  1024   // NH*VD
#define XPAD  8
#define LDX   (C_DIM + XPAD)   // 520 bf16 -> 1040B row stride (4-bank rotation per row)

typedef __attribute__((address_space(3))) float lds_f32;

// Native conversion: backend emits packed v_cvt_pk_bf16_f32.
__device__ __forceinline__ v2bf cvt2(float x, float y) {
  v2bf r; r[0] = (__bf16)x; r[1] = (__bf16)y; return r;
}

__device__ __forceinline__ float fast_tanh(float x) {
  float e = __expf(2.0f * x);
  return 1.0f - 2.0f * __builtin_amdgcn_rcpf(e + 1.0f);
}

// A-fragment 16x32 (MxK) bf16, lane: m=lane&15, hi=lane>>4.
// element e<8 -> K = kbase + hi*8 + e ; e>=8 -> K = kbase + 16 + hi*8 + (e-8).
// p points at (row base + kbase + hi*8).
__device__ __forceinline__ v16bf load_a_frag(const __bf16* p) {
  union { v16bf v; v8bf h[2]; } u;
  u.h[0] = *(const v8bf*)(p);
  u.h[1] = *(const v8bf*)(p + 16);
  return u.v;
}

// B-fragment 32x16 (KxN) bf16, lane: n=lane&15, hi=lane>>4, element e<8 -> K = kbase + hi*8 + e ; e>=8 -> K = kbase + 16 + hi*8 + (e-8).
__device__ __forceinline__ v16bf load_b_frag(const __bf16* p) {
  union { v16bf v; v8bf h[2]; } u;
  u.h[0] = *(const v8bf*)(p);
  u.h[1] = *(const v8bf*)(p + 16);
  return u.v;
}

// A-fragment built from contiguous fp32 row in global memory (converted to bf16).
__device__ __forceinline__ v16bf cvt_a_frag(const float* p /* row + kbase + hi*8 */) {
  union { v16bf v; v2bf d[8]; } u;
  v4f q0 = *(const v4f*)(p);
  v4f q1 = *(const v4f*)(p + 4);
  v4f q2 = *(const v4f*)(p + 16);
  v4f q3 = *(const v4f*)(p + 20);
  u.d[0] = cvt2(q0[0], q0[1]); u.d[1] = cvt2(q0[2], q0[3]);
  u.d[2] = cvt2(q1[0], q1[1]); u.d[3] = cvt2(q1[2], q1[3]);
  u.d[4] = cvt2(q2[0], q2[1]); u.d[5] = cvt2(q2[2], q2[3]);
  u.d[6] = cvt2(q3[0], q3[1]); u.d[7] = cvt2(q3[2], q3[3]);
  return u.v;
}

// ---------------- kernel 0: zero the output (atomics accumulate into it) ----------------
__global__ void mha_zero_out(float* out, int n) {
  int i = blockIdx.x * 256 + threadIdx.x;
  if (i < n) out[i] = 0.0f;
}

// ---------------- kernel 1: w_dot = W_k @ q per head (transposed, bf16) + bias_dot ------
__global__ void mha_prep_wdot(const float* __restrict__ Wk, const float* __restrict__ bk,
                              const float* __restrict__ q, __bf16* __restrict__ wdt,
                              float* __restrict__ bias_dot) {
  int tid = threadIdx.x;
  for (int idx = tid; idx < NH * C_DIM; idx += 256) {
    int n = idx >> 9, c = idx & (C_DIM - 1);
    const float* wrow = Wk + (size_t)c * NCOL + n * KD;
    const float* qn = q + n * KD;
    float s = 0.f;
#pragma unroll 8
    for (int k = 0; k < KD; ++k) s += wrow[k] * qn[k];
    wdt[n * C_DIM + c] = (__bf16)s;   // wdt[n][c] (B^T layout, row = output column)
  }
  if (tid < NH) {
    const float* qn = q + tid * KD;
    const float* bn = bk + tid * KD;
    float s = 0.f;
    for (int k = 0; k < KD; ++k) s += bn[k] * qn[k];
    bias_dot[tid] = s;
  }
}

// ---------------- kernel 2: Wt[ncol][c] = bf16(W_v[c][ncol]) ----------------------------
__global__ void mha_prep_wt(const float* __restrict__ Wv, __bf16* __restrict__ Wt) {
  int idx = blockIdx.x * 256 + threadIdx.x;      // 1024*512 elements, grid 2048x256
  int ncol = idx >> 9, c = idx & (C_DIM - 1);
  Wt[(size_t)ncol * C_DIM + c] = (__bf16)Wv[(size_t)c * NCOL + ncol];
}

// ---------------- kernel 3: dot[b][n][s] = x . w_dot (bf16 WMMA) ------------------------
__global__ __launch_bounds__(256)
void mha_dot(const float* __restrict__ x, const __bf16* __restrict__ wdt,
             const float* __restrict__ bias_dot, float* __restrict__ dws) {
  int b = blockIdx.x >> 5;          // 32 chunks per batch, 128 s-rows per block
  int chunk = blockIdx.x & 31;
  int wave = threadIdx.x >> 5, lane = threadIdx.x & 31;
  int m = lane & 15, hi = lane >> 4;
  int s0 = chunk * 128 + wave * 16;

  const float* arow = x + ((size_t)b * S_DIM + s0 + m) * C_DIM;
  const __bf16* brow = wdt + m * C_DIM;     // n = lane&15
  v8f acc = {};
#pragma unroll 4
  for (int kk = 0; kk < 16; ++kk) {
    v16bf a  = cvt_a_frag(arow + kk * 32 + hi * 8);
    v16bf bf = load_b_frag(brow + kk * 32 + hi * 8);
    acc = __builtin_amdgcn_wmma_f32_16x16x32_bf16(false, a, false, bf, (short)0, acc, false, false);
  }
  float bd = bias_dot[m];
  float* orow = dws + ((size_t)b * NH + m) * S_DIM + s0 + hi * 8;  // D: M=r+8*hi, N=lane&15
#pragma unroll
  for (int r = 0; r < 8; ++r) orow[r] = acc[r] + bd;
}

// ---------------- kernel 4: softmax over s per (b,n) ------------------------------------
__global__ __launch_bounds__(256)
void mha_softmax(float* __restrict__ dws) {
  int b = blockIdx.x >> 4, n = blockIdx.x & 15;
  float* p = dws + ((size_t)b * NH + n) * S_DIM;
  int tid = threadIdx.x;
  __shared__ float red[256];

  float v[16];
  float mx = -3.402823466e38f;
#pragma unroll
  for (int k = 0; k < 16; ++k) { v[k] = p[tid + k * 256]; mx = fmaxf(mx, v[k]); }
  red[tid] = mx; __syncthreads();
  for (int off = 128; off > 0; off >>= 1) {
    if (tid < off) red[tid] = fmaxf(red[tid], red[tid + off]);
    __syncthreads();
  }
  mx = red[0]; __syncthreads();

  float sum = 0.f;
#pragma unroll
  for (int k = 0; k < 16; ++k) { v[k] = __expf(v[k] - mx); sum += v[k]; }
  red[tid] = sum; __syncthreads();
  for (int off = 128; off > 0; off >>= 1) {
    if (tid < off) red[tid] += red[tid + off];
    __syncthreads();
  }
  float inv = 1.0f / red[0];
#pragma unroll
  for (int k = 0; k < 16; ++k) p[tid + k * 256] = v[k] * inv;
}

// ---------------- kernel 5: value = tanh(x@W_v+b_v), attn-weighted reduce ---------------
#define STILE 32
__global__ __launch_bounds__(256)
void mha_value(const float* __restrict__ x, const __bf16* __restrict__ Wt,
               const float* __restrict__ bv, const float* __restrict__ attn,
               float* __restrict__ out) {
  __shared__ __align__(16) __bf16 xs[STILE * LDX];   // 33,280 B
  __shared__ __align__(16) float attn_s[STILE * NH]; //  2,048 B
  int b = blockIdx.x >> 7;                           // 128 s-tiles of 32 rows per batch
  int st = blockIdx.x & 127;
  int s0 = st * STILE;
  int tid = threadIdx.x;

  // Async global->LDS copy of the attention weights (ASYNCcnt-tracked DMA path).
  for (int i = tid; i < STILE * NH; i += 256) {
    int n = i >> 5, sl = i & (STILE - 1);
    const float* gp = attn + ((size_t)b * NH + n) * S_DIM + s0 + sl;
    unsigned lds_off = (unsigned)(uintptr_t)(lds_f32*)&attn_s[sl * NH + n];
    unsigned long long ga = (unsigned long long)(uintptr_t)gp;
    asm volatile("global_load_async_to_lds_b32 %0, %1, off"
                 :: "v"(lds_off), "v"(ga) : "memory");
  }

  // Stage x tile as bf16 (pairwise hardware converts, vector loads/stores).
  const float* xb = x + ((size_t)b * S_DIM + s0) * C_DIM;
  for (int i = tid; i < STILE * (C_DIM / 2); i += 256) {   // 8192 float pairs
    int sl = i >> 8;                 // 256 pairs per row
    int c = (i & 255) * 2;
    v2f f = *(const v2f*)(xb + (size_t)sl * C_DIM + c);
    *(v2bf*)(&xs[sl * LDX + c]) = cvt2(f[0], f[1]);
  }

  asm volatile("s_wait_asynccnt 0" ::: "memory");
  __syncthreads();

  int wave = tid >> 5, lane = tid & 31;
  int m = lane & 15, hi = lane >> 4;

  // Two adjacent column-tiles per iteration: each A/B fragment feeds 2 WMMAs,
  // 4 independent accumulator chains for co-execution.
  for (int t2 = 0; t2 < 4; ++t2) {
    int j0 = wave * 8 + t2 * 2;                      // even tile index
    int head = j0 >> 2;                              // j0 and j0+1 share the head
    int col0 = j0 * 16 + m;
    int col1 = col0 + 16;
    float bv0 = bv[col0], bv1 = bv[col1];
    const __bf16* brow0 = Wt + (size_t)col0 * C_DIM;
    const __bf16* brow1 = Wt + (size_t)col1 * C_DIM;

    v8f acc00 = {}, acc10 = {}, acc01 = {}, acc11 = {};   // acc[sub][jj]
#pragma unroll 4
    for (int kk = 0; kk < 16; ++kk) {
      v16bf a0  = load_a_frag(xs + m * LDX + kk * 32 + hi * 8);
      v16bf a1  = load_a_frag(xs + (16 + m) * LDX + kk * 32 + hi * 8);
      v16bf bf0 = load_b_frag(brow0 + kk * 32 + hi * 8);
      v16bf bf1 = load_b_frag(brow1 + kk * 32 + hi * 8);
      acc00 = __builtin_amdgcn_wmma_f32_16x16x32_bf16(false, a0, false, bf0, (short)0, acc00, false, false);
      acc10 = __builtin_amdgcn_wmma_f32_16x16x32_bf16(false, a1, false, bf0, (short)0, acc10, false, false);
      acc01 = __builtin_amdgcn_wmma_f32_16x16x32_bf16(false, a0, false, bf1, (short)0, acc01, false, false);
      acc11 = __builtin_amdgcn_wmma_f32_16x16x32_bf16(false, a1, false, bf1, (short)0, acc11, false, false);
    }

    float psum0 = 0.f, psum1 = 0.f;
#pragma unroll
    for (int r = 0; r < 8; ++r) {
      int sl0 = r + hi * 8;                          // D layout: M = r + 8*hi
      int sl1 = 16 + sl0;
      float w0 = attn_s[sl0 * NH + head];
      float w1 = attn_s[sl1 * NH + head];
      psum0 += w0 * fast_tanh(acc00[r] + bv0) + w1 * fast_tanh(acc10[r] + bv0);
      psum1 += w0 * fast_tanh(acc01[r] + bv1) + w1 * fast_tanh(acc11[r] + bv1);
    }
    psum0 += __shfl_xor(psum0, 16, 32);              // combine hi/lo half-lane rows
    psum1 += __shfl_xor(psum1, 16, 32);
    if (hi == 0) {
      float* obase = out + ((size_t)b * NH + head) * VD;
      atomicAdd(obase + (col0 & (VD - 1)), psum0);
      atomicAdd(obase + (col1 & (VD - 1)), psum1);
    }
  }
}

extern "C" void kernel_launch(void* const* d_in, const int* in_sizes, int n_in,
                              void* d_out, int out_size, void* d_ws, size_t ws_size,
                              hipStream_t stream) {
  const float* x  = (const float*)d_in[0];   // kv  [32,64,64,512] -> [32,4096,512]
  const float* Wk = (const float*)d_in[1];   // [512,1024]
  const float* bk = (const float*)d_in[2];   // [1024]
  const float* Wv = (const float*)d_in[3];   // [512,1024]
  const float* bv = (const float*)d_in[4];   // [1024]
  const float* q  = (const float*)d_in[5];   // [1,1,16,64]
  float* out = (float*)d_out;                // [32,16,64]

  char* ws = (char*)d_ws;
  __bf16* wdt      = (__bf16*)(ws);                              // 16 KB
  float*  bias_dot = (float*)(ws + 16384);                       // 64 B
  __bf16* Wt       = (__bf16*)(ws + 16640);                      // 1 MB
  float*  dws      = (float*)(ws + 16640 + 1048576);             // 8 MB ([b][n][s])

  mha_zero_out<<<(out_size + 255) / 256, 256, 0, stream>>>(out, out_size);
  mha_prep_wdot<<<1, 256, 0, stream>>>(Wk, bk, q, wdt, bias_dot);
  mha_prep_wt<<<(NCOL * C_DIM) / 256, 256, 0, stream>>>(Wv, Wt);
  mha_dot<<<B_DIM * 32, 256, 0, stream>>>(x, wdt, bias_dot, dws);
  mha_softmax<<<B_DIM * NH, 256, 0, stream>>>(dws);
  mha_value<<<B_DIM * (S_DIM / STILE), 256, 0, stream>>>(x, Wt, bv, dws, out);
}
